// NeRFRenderer_33895881900669
// MI455X (gfx1250) — matrix operand order repaired and checked by reading, not compiled, not run on hardware
//
#include <hip/hip_runtime.h>

typedef __attribute__((ext_vector_type(16))) _Float16 v16h;
typedef __attribute__((ext_vector_type(8)))  float    v8f;

#define NC    64
#define NF    64
#define NALL  128
#define HID   128
#define WAVES 8

__device__ __forceinline__ void lds_fence() {
    asm volatile("s_wait_dscnt 0" ::: "memory");
}

// inclusive multiply-scan across the wave (32 lanes)
__device__ __forceinline__ float iscan_mul(float p, int lane) {
#pragma unroll
    for (int d = 1; d < 32; d <<= 1) {
        float t = __shfl_up(p, d);
        if (lane >= d) p *= t;
    }
    return p;
}

// inclusive add-scan across the wave
__device__ __forceinline__ float iscan_add(float p, int lane) {
#pragma unroll
    for (int d = 1; d < 32; d <<= 1) {
        float t = __shfl_up(p, d);
        if (lane >= d) p += t;
    }
    return p;
}

// full-wave sum reduction (result in all lanes)
__device__ __forceinline__ float wred_add(float x) {
#pragma unroll
    for (int d = 16; d > 0; d >>= 1) x += __shfl_xor(x, d);
    return x;
}

// Run NT tiles of 16 samples through layer1 (a+z*b, relu) feeding
// v_wmma_f32_16x16x32_f16 (K=128 as 4 chunks of 32), scatter rgb+sigma to LDS.
template <int NT>
__device__ __forceinline__ void mlp_tiles(const float2* __restrict__ abw,
                                          const float* __restrict__ zbuf,
                                          const v16h Bf[4], float b2v,
                                          float (*rgbs)[4], int lane) {
    const int m  = lane & 15;   // sample row within tile / output column
    const int hi = lane >> 4;   // K-half selector for A fragment

    float zt[NT];
#pragma unroll
    for (int t = 0; t < NT; ++t) zt[t] = zbuf[16 * t + m];

    v8f acc[NT];
#pragma unroll
    for (int t = 0; t < NT; ++t) {
#pragma unroll
        for (int v = 0; v < 8; ++v) acc[t][v] = b2v;   // seed with b2
    }

#pragma unroll
    for (int c = 0; c < 4; ++c) {
        // lane's 16 (a,b) pairs for this K-chunk, per 16-bit A layout:
        // element e -> k = 32c + 16*(e>>3) + 8*hi + (e&7)
        float2 abv[16];
#pragma unroll
        for (int e = 0; e < 16; ++e) {
            int k = 32 * c + ((e >> 3) << 4) + (hi << 3) + (e & 7);
            abv[e] = abw[k];
        }
#pragma unroll
        for (int t = 0; t < NT; ++t) {
            v16h A;
#pragma unroll
            for (int e = 0; e < 16; ++e) {
                float h = fmaf(zt[t], abv[e].y, abv[e].x);
                h = fmaxf(h, 0.0f);
                A[e] = (_Float16)h;
            }
            acc[t] = __builtin_amdgcn_wmma_f32_16x16x32_f16(
                false, A, false, Bf[c], (short)0, acc[t], false, false);
        }
    }

    // C layout: VGPR v, lanes 0-15 -> M=v, N=lane; lanes 16-31 -> M=v+8, N=lane-16
    if (m < 4) {
#pragma unroll
        for (int t = 0; t < NT; ++t) {
#pragma unroll
            for (int v = 0; v < 8; ++v) {
                rgbs[16 * t + v + 8 * hi][m] = acc[t][v];
            }
        }
    }
}

__global__ __launch_bounds__(256) void nerf_fused_kernel(
    const float* __restrict__ rays, const float* __restrict__ uc,
    const float* __restrict__ uf, const float* __restrict__ uj,
    const float* __restrict__ W1, const float* __restrict__ b1,
    const float* __restrict__ W2, const float* __restrict__ b2,
    float* __restrict__ out, int R) {
    __shared__ float2 s_ab[WAVES][HID];                     // (a,b) per hidden unit
    __shared__ __align__(16) float s_z[WAVES][NALL];        // z_coarse | z_fine
    __shared__ __align__(16) float s_zs[WAVES][NALL];       // sorted z_all
    __shared__ float s_cdf[WAVES][NC + 1];
    __shared__ __align__(16) float s_rgbs[WAVES][NALL][4];  // MLP outputs

    const int lane = threadIdx.x & 31;
    const int w    = threadIdx.x >> 5;
    const int r    = blockIdx.x * WAVES + w;
    if (r >= R) return;   // wave-uniform: EXEC stays all-ones for WMMA

    __builtin_prefetch(uf + r * NF, 0, 0);
    __builtin_prefetch(uj + r * NF, 0, 0);

    // ---- ray data ----
    const float4* rp = (const float4*)(rays + r * 8);
    float4 ra = rp[0], rb = rp[1];
    const float ox = ra.x, oy = ra.y, oz = ra.z;
    const float dx = ra.w, dy = rb.x, dz = rb.y;
    const float near = rb.z, far = rb.w, span = far - near;

    const int m  = lane & 15;
    const int hi = lane >> 4;

    // ---- B fragments: W2 (128x4) padded to 128x16, f16 ----
    // B layout (matches A): element e -> k = 32c + 16*(e>>3) + 8*hi + (e&7), col n = lane%16
    v16h Bf[4];
    const float b2v = (m < 4) ? b2[m] : 0.0f;
#pragma unroll
    for (int c = 0; c < 4; ++c) {
#pragma unroll
        for (int e = 0; e < 16; ++e) {
            int k   = 32 * c + 16 * (e >> 3) + 8 * hi + (e & 7);
            float v = W2[k * 4 + ((m < 4) ? m : 0)];
            Bf[c][e] = (_Float16)((m < 4) ? v : 0.0f);
        }
    }

    // ---- per-ray layer-1 folding: a[k] = o.W1col + b1, b[k] = d.W1col ----
#pragma unroll
    for (int i = 0; i < 4; ++i) {
        int k    = lane + 32 * i;
        float c0 = W1[k], c1 = W1[HID + k], c2 = W1[2 * HID + k];
        float a  = fmaf(ox, c0, fmaf(oy, c1, fmaf(oz, c2, b1[k])));
        float b  = fmaf(dx, c0, fmaf(dy, c1, dz * c2));
        s_ab[w][k] = make_float2(a, b);
    }

    // ---- coarse z: z_steps = (i+u)/64 ----
#pragma unroll
    for (int j = 0; j < 2; ++j) {
        int s   = 2 * lane + j;
        float u = uc[r * NC + s];
        s_z[w][s] = fmaf(span, (s + u) * (1.0f / NC), near);
    }
    lds_fence();

    // ---- coarse MLP: 4 tiles of 16 samples ----
    mlp_tiles<4>(s_ab[w], s_z[w], Bf, b2v, s_rgbs[w], lane);
    lds_fence();

    // ---- coarse composite (2 samples/lane, s = 2*lane+j) ----
    const int s0 = 2 * lane, s1 = s0 + 1;
    float z0 = s_z[w][s0], z1 = s_z[w][s1];
    float z2 = (lane < 31) ? s_z[w][s1 + 1] : far;
    float4 o0 = *(const float4*)s_rgbs[w][s0];
    float4 o1 = *(const float4*)s_rgbs[w][s1];
    float a0 = 1.0f - __expf(-(z1 - z0) * fmaxf(o0.w, 0.0f));
    float a1 = 1.0f - __expf(-(z2 - z1) * fmaxf(o1.w, 0.0f));
    float g0 = 1.0f - a0 + 1e-10f;
    float g1 = 1.0f - a1 + 1e-10f;
    float P   = iscan_mul(g0 * g1, lane);
    float Pex = __shfl_up(P, 1);
    if (lane == 0) Pex = 1.0f;
    float w0s = a0 * Pex;
    float w1s = a1 * Pex * g0;
    float rc_r  = wred_add(w0s * o0.x + w1s * o1.x);
    float rc_g  = wred_add(w0s * o0.y + w1s * o1.y);
    float rc_b  = wred_add(w0s * o0.z + w1s * o1.z);
    float dep_c = wred_add(w0s * z0 + w1s * z1);

    // ---- pdf / cdf ----
    float pw0 = w0s + 1e-5f, pw1 = w1s + 1e-5f;
    float inv = 1.0f / wred_add(pw0 + pw1);
    float S   = iscan_add((pw0 + pw1) * inv, lane);   // inclusive over lanes
    s_cdf[w][s1 + 1] = S;
    s_cdf[w][s0 + 1] = S - pw1 * inv;
    if (lane == 0) s_cdf[w][0] = 0.0f;
    lds_fence();

    // ---- inverse-CDF sampling: searchsorted(cdf[0..64], u, right) ----
#pragma unroll
    for (int j = 0; j < 2; ++j) {
        int s   = 2 * lane + j;
        float u = uf[r * NF + s];
        int lo = 0, hicnt = NC + 1;
        while (lo < hicnt) {
            int mid = (lo + hicnt) >> 1;
            if (s_cdf[w][mid] <= u) lo = mid + 1; else hicnt = mid;
        }
        float fi = fmaxf((float)lo - 1.0f, 0.0f);
        float t  = (fi + uj[r * NF + s]) * (1.0f / NC);
        s_z[w][NC + s] = fmaf(span, t, near);
    }
    lds_fence();

    // ---- sort z_all (128) by counting rank; uniform LDS broadcast reads ----
    float vals[4];
    int   rnk[4];
#pragma unroll
    for (int j = 0; j < 4; ++j) { vals[j] = s_z[w][lane + 32 * j]; rnk[j] = 0; }
    for (int q = 0; q < NALL; ++q) {
        float zq = s_z[w][q];
#pragma unroll
        for (int j = 0; j < 4; ++j) {
            int e = lane + 32 * j;
            rnk[j] += (zq < vals[j]) || (zq == vals[j] && q < e);
        }
    }
#pragma unroll
    for (int j = 0; j < 4; ++j) s_zs[w][rnk[j]] = vals[j];
    lds_fence();

    // ---- fine MLP: 8 tiles of 16 samples ----
    mlp_tiles<8>(s_ab[w], s_zs[w], Bf, b2v, s_rgbs[w], lane);
    lds_fence();

    // ---- fine composite (4 samples/lane, s = 4*lane+j) ----
    float zf[5];
#pragma unroll
    for (int j = 0; j < 4; ++j) zf[j] = s_zs[w][4 * lane + j];
    zf[4] = (lane < 31) ? s_zs[w][4 * lane + 4] : far;
    float al[4], gg[4];
    float4 oo[4];
#pragma unroll
    for (int j = 0; j < 4; ++j) {
        oo[j]     = *(const float4*)s_rgbs[w][4 * lane + j];
        float dlt = zf[j + 1] - zf[j];             // last sample: far - z[127]
        al[j] = 1.0f - __expf(-dlt * fmaxf(oo[j].w, 0.0f));
        gg[j] = 1.0f - al[j] + 1e-10f;
    }
    float Pf   = iscan_mul(gg[0] * gg[1] * gg[2] * gg[3], lane);
    float Pex2 = __shfl_up(Pf, 1);
    if (lane == 0) Pex2 = 1.0f;
    float Tj = Pex2;
    float rf_r = 0.0f, rf_g = 0.0f, rf_b = 0.0f, dep_f = 0.0f;
#pragma unroll
    for (int j = 0; j < 4; ++j) {
        float wj = al[j] * Tj;
        rf_r += wj * oo[j].x;
        rf_g += wj * oo[j].y;
        rf_b += wj * oo[j].z;
        dep_f += wj * zf[j];
        Tj *= gg[j];
    }
    rf_r  = wred_add(rf_r);
    rf_g  = wred_add(rf_g);
    rf_b  = wred_add(rf_b);
    dep_f = wred_add(dep_f);

    if (lane == 0) {
        float4* op = (float4*)(out + r * 8);
        op[0] = make_float4(rc_r, rc_g, rc_b, dep_c);
        op[1] = make_float4(rf_r, rf_g, rf_b, dep_f);
    }
}

extern "C" void kernel_launch(void* const* d_in, const int* in_sizes, int n_in,
                              void* d_out, int out_size, void* d_ws, size_t ws_size,
                              hipStream_t stream) {
    (void)n_in; (void)out_size; (void)d_ws; (void)ws_size;
    const float* rays = (const float*)d_in[0];
    const float* uc   = (const float*)d_in[1];
    const float* uf   = (const float*)d_in[2];
    const float* uj   = (const float*)d_in[3];
    const float* W1   = (const float*)d_in[4];
    const float* b1   = (const float*)d_in[5];
    const float* W2   = (const float*)d_in[6];
    const float* b2   = (const float*)d_in[7];
    float* out        = (float*)d_out;

    const int R = in_sizes[0] / 8;                 // rays is (R, 8)
    dim3 block(32 * WAVES);                        // 8 waves, one ray each
    dim3 grid((R + WAVES - 1) / WAVES);
    nerf_fused_kernel<<<grid, block, 0, stream>>>(rays, uc, uf, uj, W1, b1, W2,
                                                  b2, out, R);
}
